// GraphWavenet_65077344469505
// MI455X (gfx1250) — matrix operand order repaired and checked by reading, not compiled, not run on hardware
//
#include <hip/hip_runtime.h>
#include <math.h>

// ---------------------------------------------------------------------------
// Graph-WaveNet forward, MI455X (gfx1250, wave32, WMMA).
//
// Key optimization: output depends only on t in [1008,1024) (receptive field
// of the last-step skip readout is 15), so we compute the 16-step dependency
// cone only (64x less work than the reference). Dilated temporal convs are
// f16 WMMA GEMMs: [6656,384]@[384,128] with A rows gathered from taps t and
// t-d; fused bias + tanh*sigmoid gating in the epilogue.
// ---------------------------------------------------------------------------

typedef _Float16 f16;
typedef __attribute__((ext_vector_type(16))) _Float16 v16h;
typedef __attribute__((ext_vector_type(8)))  _Float16 v8h;
typedef __attribute__((ext_vector_type(8)))  float    v8f;

#define BATCH 16
#define TWIN  16                  // timestep window [1008,1024)
#define T0    1008
#define TFULL 1024
#define NJ    26                  // joints
#define NF    64                  // filters
#define NSKIP 256
#define KEXT  192                 // 3*NF stacked features per tap
#define ROWS_PER_B (TWIN*NJ)      // 416 (multiple of 16 -> tiles never span b)
#define NROWS (BATCH*ROWS_PER_B)  // 6656

union AFrag { v16h v; v8h h[2]; };

// --- Preformat conv weights [3,2,64,128] f32 -> WMMA B fragments (f16) -----
// Layout: Bf[blk][tap][kc(6)][ntile(8)][lane(32)][16 halves]
// B 32x16 K-striping: lanes 0-15 hold K 0..7,16..23 (col = lane), lanes 16-31
// hold K 8..15,24..31 (col = lane-16); halves 0-7 then 8-15 of each lane.
__global__ void k_wfmt(const float* __restrict__ w0, const float* __restrict__ w1,
                       const float* __restrict__ w2, const float* __restrict__ w3,
                       f16* __restrict__ Bf)
{
  int tid = blockIdx.x * blockDim.x + threadIdx.x;
  if (tid >= 4*2*6*8*32*16) return;
  int idx = tid;
  int h    = idx & 15;  idx >>= 4;
  int lane = idx & 31;  idx >>= 5;
  int nt   = idx & 7;   idx >>= 3;
  int kc   = idx % 6;   idx /= 6;
  int tap  = idx & 1;   idx >>= 1;
  int blk  = idx;
  const float* w = (blk == 0) ? w0 : (blk == 1) ? w1 : (blk == 2) ? w2 : w3;
  int Kc = (h < 8) ? ((lane < 16) ? h : 8 + h) : ((lane < 16) ? 8 + h : 16 + h);
  int Kg = kc*32 + Kc;                 // 0..191: stacked feature index
  int hs = Kg >> 6, c = Kg & 63;       // stack row (self/parent/gp), channel
  int o  = nt*16 + (lane & 15);        // output channel 0..127
  float val = w[((hs*2 + tap)*64 + c)*128 + o];
  Bf[(size_t)blk*49152 + ((((tap*6 + kc)*8 + nt)*32 + lane)*16 + h)] = (f16)val;
}

// --- Embedding on the 16-step window: X[(b,tau,j),f] = xin @ emb_w + emb_b --
__global__ void k_embed(const float* __restrict__ xin, const float* __restrict__ ew,
                        const float* __restrict__ eb, float* __restrict__ X)
{
  int tid = blockIdx.x * blockDim.x + threadIdx.x;
  if (tid >= NROWS*NF) return;
  int f  = tid & 63;
  int r  = tid >> 6;
  int j  = r % NJ;
  int bt = r / NJ;
  int tau = bt & 15, b = bt >> 4;
  int src = ((b*TFULL + T0 + tau)*NJ + j)*3;
  float v = eb[f];
  v += xin[src+0]*ew[0*64+f];
  v += xin[src+1]*ew[1*64+f];
  v += xin[src+2]*ew[2*64+f];
  X[tid] = v;
}

// --- Build E[(b,tau,j), 0:192] = {x, P@x, P@P@x} as f16 (one WG per (b,tau))
__global__ __launch_bounds__(256) void k_ebuild(const float* __restrict__ X,
    const float* __restrict__ P, f16* __restrict__ E)
{
  __shared__ float xs[NJ*NF];
  __shared__ float ps[NJ*NF];
  __shared__ float Ps[NJ*NJ];
  int bt = blockIdx.x, tid = threadIdx.x;
  const float* xb = X + (size_t)bt * (NJ*NF);
  f16* Eb = E + (size_t)bt * (NJ*KEXT);
  for (int i = tid; i < NJ*NJ; i += 256) Ps[i] = P[i];
  for (int i = tid; i < NJ*NF; i += 256) {
    float v = xb[i];
    xs[i] = v;
    Eb[(i >> 6)*KEXT + (i & 63)] = (f16)v;
  }
  __syncthreads();
  for (int i = tid; i < NJ*NF; i += 256) {
    int u = i >> 6, f = i & 63;
    float acc = 0.f;
    #pragma unroll
    for (int v = 0; v < NJ; ++v) acc += Ps[u*NJ+v] * xs[v*64 + f];
    ps[i] = acc;
    Eb[u*KEXT + 64 + f] = (f16)acc;
  }
  __syncthreads();
  for (int i = tid; i < NJ*NF; i += 256) {
    int u = i >> 6, f = i & 63;
    float acc = 0.f;
    #pragma unroll
    for (int v = 0; v < NJ; ++v) acc += Ps[u*NJ+v] * ps[v*64 + f];
    Eb[u*KEXT + 128 + f] = (f16)acc;
  }
}

// --- Dilated causal conv as WMMA GEMM + fused gating --------------------
// One wave per 16-row M-tile; full N=128 (8 C fragments). K = 2 taps x 192.
// Tap0 reads rows shifted by -NJ*dil (zeroed per-lane where tau < dil).
__global__ __launch_bounds__(32) void k_conv_wmma(const f16* __restrict__ E,
    const f16* __restrict__ Bf, const float* __restrict__ cb,
    float* __restrict__ XG, int dil)
{
  int lane = threadIdx.x & 31;
  int rowbase = blockIdx.x * 16;
  int lrow = rowbase + (lane & 15);          // A layout: lane&15 selects row
  int tau  = (lrow % ROWS_PER_B) / NJ;
  int offA = (lane < 16) ? 0 : 8;            // A layout: K-chunk offset

  v8f acc[8] = {};

  #pragma unroll
  for (int tap = 0; tap < 2; ++tap) {
    int arow   = (tap == 1) ? lrow : (lrow - NJ*dil);
    bool valid = (tap == 1) || (tau >= dil);
    #pragma unroll
    for (int kc = 0; kc < 6; ++kc) {
      AFrag a = {};
      if (valid) {
        const f16* ap = E + (size_t)arow*KEXT + kc*32 + offA;
        a.h[0] = *(const v8h*)ap;            // K offA..offA+7
        a.h[1] = *(const v8h*)(ap + 16);     // K offA+16..offA+23
      }
      const f16* bp = Bf + (size_t)(((tap*6 + kc)*8)*32 + lane)*16;
      #pragma unroll
      for (int nt = 0; nt < 8; ++nt) {
        v16h b = *(const v16h*)(bp + (size_t)nt*(32*16));
        acc[nt] = __builtin_amdgcn_wmma_f32_16x16x32_f16(
            false, a.v, false, b, (short)0, acc[nt], false, false);
      }
    }
  }

  // Epilogue: bias + tanh(act)*sigmoid(gate). act = cols 0..63 (tiles 0..3),
  // gate = cols 64..127 (tiles 4..7). C layout: lane<16 -> (M=v, N=lane),
  // lane>=16 -> (M=v+8, N=lane-16).
  #pragma unroll
  for (int ta = 0; ta < 4; ++ta) {
    int col = ta*16 + (lane & 15);
    float ba = cb[col], bg = cb[64 + col];
    #pragma unroll
    for (int v = 0; v < 8; ++v) {
      int row = rowbase + v + ((lane < 16) ? 0 : 8);
      float aval = acc[ta][v] + ba;
      float gval = acc[ta+4][v] + bg;
      XG[(size_t)row*NF + col] = tanhf(aval) * (1.f / (1.f + __expf(-gval)));
    }
  }
}

// --- Skip accumulation: SK[b,j,:] (+)= gated[b, tau=15, j, :] @ skip_w + b --
__global__ void k_skip(const float* __restrict__ XG, const float* __restrict__ sw,
                       const float* __restrict__ sb, float* __restrict__ SK, int init)
{
  int tid = blockIdx.x * blockDim.x + threadIdx.x;
  if (tid >= BATCH*NJ*NSKIP) return;
  int s = tid & 255;
  int j = (tid >> 8) % NJ;
  int b = tid / (256*NJ);
  const float* xr = XG + (size_t)(((b*TWIN + (TWIN-1))*NJ + j))*NF;
  float acc = sb[s];
  #pragma unroll
  for (int f = 0; f < NF; ++f) acc += xr[f] * sw[f*NSKIP + s];
  if (init) SK[tid] = acc; else SK[tid] += acc;
}

// --- GCN ((P+I)@xg)@gcn_w + residual; writes X in place (one WG per (b,tau))
__global__ __launch_bounds__(256) void k_gcn(const float* __restrict__ XG,
    const float* __restrict__ P, const float* __restrict__ gw,
    float* __restrict__ X)
{
  __shared__ float xs[NJ*NF], tm[NJ*NF], Ps[NJ*NJ];
  int bt = blockIdx.x, tid = threadIdx.x;
  const float* xgb = XG + (size_t)bt*(NJ*NF);
  float* xb = X + (size_t)bt*(NJ*NF);
  for (int i = tid; i < NJ*NJ; i += 256) Ps[i] = P[i];
  for (int i = tid; i < NJ*NF; i += 256) xs[i] = xgb[i];
  __syncthreads();
  for (int i = tid; i < NJ*NF; i += 256) {
    int u = i >> 6, f = i & 63;
    float acc = xs[u*64 + f];                 // identity (A = P + I)
    #pragma unroll
    for (int v = 0; v < NJ; ++v) acc += Ps[u*NJ+v] * xs[v*64 + f];
    tm[i] = acc;
  }
  __syncthreads();
  for (int i = tid; i < NJ*NF; i += 256) {
    int u = i >> 6, o = i & 63;
    float acc = xb[i];                        // residual
    #pragma unroll
    for (int f = 0; f < NF; ++f) acc += tm[u*64 + f] * gw[f*64 + o];
    xb[i] = acc;
  }
}

// --- Output head: relu(skip) -> relu(@W1+b1) -> @W2+b2 (one WG per (b,j)) --
__global__ __launch_bounds__(256) void k_head(const float* __restrict__ SK,
    const float* __restrict__ w1, const float* __restrict__ b1,
    const float* __restrict__ w2, const float* __restrict__ b2,
    float* __restrict__ out)
{
  __shared__ float ss[NSKIP], hs[NSKIP];
  int bj = blockIdx.x, t = threadIdx.x;
  float sv = SK[(size_t)bj*NSKIP + t];
  ss[t] = sv > 0.f ? sv : 0.f;
  __syncthreads();
  float acc = b1[t];
  for (int k = 0; k < NSKIP; ++k) acc += ss[k] * w1[k*NSKIP + t];
  hs[t] = acc > 0.f ? acc : 0.f;
  __syncthreads();
  if (t < 3) {
    float o = b2[t];
    for (int k = 0; k < NSKIP; ++k) o += hs[k] * w2[k*3 + t];
    out[bj*3 + t] = o;
  }
}

// ---------------------------------------------------------------------------
// Input order: setup_inputs() dict insertion order, recursively:
//  0:x 1:adj 2:parent_mat 3:kin_mat 4:emb_w 5:emb_b 6:out_w1 7:out_b1
//  8:out_w2 9:out_b2, then blocks[i]: 10+5i:conv_w +1:conv_b +2:skip_w
//  +3:skip_b +4:gcn_w
// ---------------------------------------------------------------------------
extern "C" void kernel_launch(void* const* d_in, const int* in_sizes, int n_in,
                              void* d_out, int out_size, void* d_ws, size_t ws_size,
                              hipStream_t stream)
{
  (void)in_sizes; (void)n_in; (void)out_size; (void)ws_size;
  const float* xin = (const float*)d_in[0];
  const float* P   = (const float*)d_in[2];   // parent_mat[0]
  const float* ew  = (const float*)d_in[4];
  const float* eb  = (const float*)d_in[5];
  const float* ow1 = (const float*)d_in[6];
  const float* ob1 = (const float*)d_in[7];
  const float* ow2 = (const float*)d_in[8];
  const float* ob2 = (const float*)d_in[9];
  const float *convw[4], *convb[4], *skw[4], *skb[4], *gw[4];
  for (int i = 0; i < 4; ++i) {
    convw[i] = (const float*)d_in[10 + 5*i + 0];
    convb[i] = (const float*)d_in[10 + 5*i + 1];
    skw[i]   = (const float*)d_in[10 + 5*i + 2];
    skb[i]   = (const float*)d_in[10 + 5*i + 3];
    gw[i]    = (const float*)d_in[10 + 5*i + 4];
  }

  // Workspace carve-up (all 32B aligned), ~6.8 MB total:
  char* ws = (char*)d_ws;
  float* X  = (float*)(ws + 0);          // 6656*64 f32   (1,703,936 B)
  float* XG = (float*)(ws + 1703936);    // 6656*64 f32   (1,703,936 B)
  f16*   E  = (f16*)  (ws + 3407872);    // 6656*192 f16  (2,555,904 B)
  f16*   Bf = (f16*)  (ws + 5963776);    // 4*49152 f16   (  393,216 B)
  float* SK = (float*)(ws + 6356992);    // 16*26*256 f32 (  425,984 B)

  k_wfmt<<<768, 256, 0, stream>>>(convw[0], convw[1], convw[2], convw[3], Bf);
  k_embed<<<(NROWS*NF)/256, 256, 0, stream>>>(xin, ew, eb, X);

  const int dil[4] = {1, 2, 4, 8};
  for (int i = 0; i < 4; ++i) {
    k_ebuild<<<BATCH*TWIN, 256, 0, stream>>>(X, P, E);
    k_conv_wmma<<<NROWS/16, 32, 0, stream>>>(E, Bf + (size_t)i*49152,
                                             convb[i], XG, dil[i]);
    k_skip<<<(BATCH*NJ*NSKIP)/256, 256, 0, stream>>>(XG, skw[i], skb[i], SK,
                                                     i == 0 ? 1 : 0);
    if (i < 3)
      k_gcn<<<BATCH*TWIN, 256, 0, stream>>>(XG, P, gw[i], X);
  }
  k_head<<<BATCH*NJ, 256, 0, stream>>>(SK, ow1, ob1, ow2, ob2, (float*)d_out);
}
